// rnn_decoder_83528523972882
// MI455X (gfx1250) — matrix operand order repaired and checked by reading, not compiled, not run on hardware
//
#include <hip/hip_runtime.h>

// ---------------------------------------------------------------------------
// CDNA5 (gfx1250) WMMA types & helpers
// ---------------------------------------------------------------------------
typedef __attribute__((ext_vector_type(16))) __bf16 v16bf;
typedef __attribute__((ext_vector_type(8)))  float  v8f;

struct alignas(16) U4 { unsigned int x, y, z, w; };
union BFrag { v16bf v; U4 q[2]; };

__device__ __forceinline__ unsigned short f2b(float f) {
    unsigned int u = __float_as_uint(f);
    unsigned int r = u + 0x7FFFu + ((u >> 16) & 1u);   // round-to-nearest-even
    return (unsigned short)(r >> 16);
}

// A-fragment: 16x32 bf16 (M x K).  lane 0-15: M=lane, K=0..15;
// lane 16-31: M=lane-16, K=16..31.  Two aligned 16B loads.
__device__ __forceinline__ v16bf load_a_frag(const unsigned short* __restrict__ X,
                                             int ld, int m0, int k0, int lane) {
    int m = lane & 15, kh = lane >> 4;
    const unsigned short* p = X + (long)(m0 + m) * ld + (k0 + 16 * kh);
    BFrag f;
    f.q[0] = *(const U4*)(p);
    f.q[1] = *(const U4*)(p + 8);
    return f.v;
}

// B-fragment from LDS row-major tile (row = output col n, padded row stride):
// lane 0-15: N=lane, K=kk..kk+15;  lane 16-31: N=lane-16, K=kk+16..kk+31.
__device__ __forceinline__ v16bf load_b_frag_lds(const unsigned short* p) {
    BFrag f;
    f.q[0] = *(const U4*)(p);
    f.q[1] = *(const U4*)(p + 8);
    return f.v;
}

__device__ __forceinline__ v8f wmma_bf16(v16bf a, v16bf b, v8f c) {
    return __builtin_amdgcn_wmma_f32_16x16x32_bf16(false, a, false, b,
                                                   (short)0, c, false, false);
}

__device__ __forceinline__ float sigm(float x) { return 1.0f / (1.0f + __expf(-x)); }

// ---------------------------------------------------------------------------
// Fused LSTM cell.  One block per 16-wide hidden tile (grid = H/16 = 64).
// 16 waves = (4 levels x 4 batch tiles) share the gate-weight tile, staged
// through LDS in double-buffered K-chunks of 128 (global->reg prefetch,
// WMMA from LDS, reg->LDS store, one barrier per chunk).
// Per K-step: 1 A-fragment + 4 gate B-fragments preloaded, then 4 independent
// WMMAs issued back-to-back; sigmoid/tanh cell update in-register.
// ---------------------------------------------------------------------------
#define CELL_LDW 136   // 128 + 8 pad (272B row stride -> bank rotation)

__global__ __launch_bounds__(512) void lstm_cell_kernel(
    const unsigned short* __restrict__ X, long xls, int Kx,
    const unsigned short* __restrict__ Hin,
    const unsigned short* __restrict__ Wih,
    const unsigned short* __restrict__ Whh,
    const float* __restrict__ bias,
    const float* __restrict__ Cin,
    float* __restrict__ Cout,
    float* __restrict__ Hf,
    unsigned short* __restrict__ Hb,
    long sls)
{
    const int H = 1024;
    __shared__ __attribute__((aligned(16))) unsigned short lds[2][64 * CELL_LDW];

    int tid = threadIdx.x;
    int lane = tid & 31, wv = tid >> 5;
    int l = wv >> 2, m0 = (wv & 3) << 4;
    int n0 = blockIdx.x << 4;
    int n = lane & 15, kh = lane >> 4;

    const unsigned short* Xl = X + (long)l * xls;
    const unsigned short* Hl = Hin + (long)l * sls;

    // staging map: 64 rows (gate*16+row) x 128 k, 16 bf16 per thread
    int sr = tid >> 3;               // 0..63
    int sk = (tid & 7) << 4;         // 0,16,...,112
    long srow = (long)((sr >> 4) * H + n0 + (sr & 15));

    int nch = (Kx + H) >> 7;         // K-chunks of 128 (Kx, H multiples of 128)
    U4 s0, s1;
    {
        const unsigned short* src = Wih + srow * Kx + sk;
        s0 = *(const U4*)src; s1 = *(const U4*)(src + 8);
        U4* d = (U4*)&lds[0][sr * CELL_LDW + sk];
        d[0] = s0; d[1] = s1;
    }
    __syncthreads();

    v8f acc[4];
#pragma unroll
    for (int g = 0; g < 4; ++g)
#pragma unroll
        for (int v = 0; v < 8; ++v) acc[g][v] = 0.0f;

    for (int c = 0; c < nch; ++c) {
        int cur = c & 1;
        if (c + 1 < nch) {           // prefetch next weight chunk into registers
            int k0 = (c + 1) << 7;
            const unsigned short* src = (k0 < Kx)
                ? Wih + srow * Kx + k0 + sk
                : Whh + srow * H + (k0 - Kx) + sk;
            s0 = *(const U4*)src; s1 = *(const U4*)(src + 8);
        }
        int k0 = c << 7;
        const unsigned short* Ap; int lda, kb;
        if (k0 < Kx) { Ap = Xl; lda = Kx; kb = k0; }
        else         { Ap = Hl; lda = H;  kb = k0 - Kx; }
#pragma unroll
        for (int kk = 0; kk < 128; kk += 32) {
            const unsigned short* bp = &lds[cur][n * CELL_LDW + kk + 16 * kh];
            v16bf a  = load_a_frag(Ap, lda, m0, kb + kk, lane);
            v16bf b0 = load_b_frag_lds(bp);
            v16bf b1 = load_b_frag_lds(bp + 16 * CELL_LDW);
            v16bf b2 = load_b_frag_lds(bp + 32 * CELL_LDW);
            v16bf b3 = load_b_frag_lds(bp + 48 * CELL_LDW);
            acc[0] = wmma_bf16(a, b0, acc[0]);
            acc[1] = wmma_bf16(a, b1, acc[1]);
            acc[2] = wmma_bf16(a, b2, acc[2]);
            acc[3] = wmma_bf16(a, b3, acc[3]);
        }
        if (c + 1 < nch) {
            U4* d = (U4*)&lds[cur ^ 1][sr * CELL_LDW + sk];
            d[0] = s0; d[1] = s1;
        }
        __syncthreads();
    }

    int col = n0 + n;
    long lo = (long)l * sls;
#pragma unroll
    for (int v = 0; v < 8; ++v) {
        int row = m0 + kh * 8 + v;
        long o = lo + (long)row * H + col;
        float iv = sigm(acc[0][v] + bias[col]);
        float fv = sigm(acc[1][v] + bias[H + col]);
        float gv = tanhf(acc[2][v] + bias[2 * H + col]);
        float ov = sigm(acc[3][v] + bias[3 * H + col]);
        float cn = fv * Cin[o] + iv * gv;
        float hn = ov * tanhf(cn);
        Cout[o] = cn;
        Hf[o]   = hn;
        Hb[o]   = f2b(hn);
    }
}

// ---------------------------------------------------------------------------
// Generic WMMA GEMM: Y[l] = act(concat(A0[l],A1[l]) @ W^T + bias), N = 1024.
// One block per n-tile (grid = 64); 16 waves = (4 levels x 4 batch tiles)
// share the 16 weight rows, staged via double-buffered LDS chunks of K=512.
// Two independent accumulators break the WMMA dependence chain; 2 A + 2 B
// fragments preloaded per unrolled K-step.
// ---------------------------------------------------------------------------
#define GEMM_LDW 520   // 512 + 8 pad

__global__ __launch_bounds__(512) void gemm_kernel(
    const unsigned short* __restrict__ A0, long a0ls, int lda0, int K0,
    const unsigned short* __restrict__ A1, long a1ls, int lda1, int K1,
    const unsigned short* __restrict__ W, int ldw,
    const float* __restrict__ bias, int act,
    float* __restrict__ oF0, long f0ls,
    float* __restrict__ oF1, long f1ls,
    unsigned short* __restrict__ oB0, long b0ls,
    unsigned short* __restrict__ oB1, long b1ls)
{
    __shared__ __attribute__((aligned(16))) unsigned short lds[2][16 * GEMM_LDW];

    int tid = threadIdx.x;
    int lane = tid & 31, wv = tid >> 5;
    int l = wv >> 2, m0 = (wv & 3) << 4;
    int n0 = blockIdx.x << 4;
    int n = lane & 15, kh = lane >> 4;

    const unsigned short* A0l = A0 + (long)l * a0ls;
    const unsigned short* A1l = A1 + (long)l * a1ls;

    int K = K0 + K1;
    int nch = K >> 9;                // chunks of 512
    int sr = tid >> 5;               // 0..15
    int sk = (tid & 31) << 4;        // 0..496
    const unsigned short* wrow = W + (long)(n0 + sr) * ldw;

    U4 s0, s1;
    {
        const unsigned short* src = wrow + sk;
        s0 = *(const U4*)src; s1 = *(const U4*)(src + 8);
        U4* d = (U4*)&lds[0][sr * GEMM_LDW + sk];
        d[0] = s0; d[1] = s1;
    }
    __syncthreads();

    v8f accA, accB;
#pragma unroll
    for (int v = 0; v < 8; ++v) { accA[v] = 0.0f; accB[v] = 0.0f; }

    for (int c = 0; c < nch; ++c) {
        int cur = c & 1;
        if (c + 1 < nch) {
            const unsigned short* src = wrow + ((c + 1) << 9) + sk;
            s0 = *(const U4*)src; s1 = *(const U4*)(src + 8);
        }
        int k0 = c << 9;
        const unsigned short* Ap; int lda, kb;
        if (k0 < K0) { Ap = A0l; lda = lda0; kb = k0; }
        else         { Ap = A1l; lda = lda1; kb = k0 - K0; }
#pragma unroll
        for (int kk = 0; kk < 512; kk += 64) {
            const unsigned short* bp = &lds[cur][n * GEMM_LDW + kk + 16 * kh];
            v16bf a0 = load_a_frag(Ap, lda, m0, kb + kk, lane);
            v16bf a1 = load_a_frag(Ap, lda, m0, kb + kk + 32, lane);
            v16bf b0 = load_b_frag_lds(bp);
            v16bf b1 = load_b_frag_lds(bp + 32);
            accA = wmma_bf16(a0, b0, accA);
            accB = wmma_bf16(a1, b1, accB);
        }
        if (c + 1 < nch) {
            U4* d = (U4*)&lds[cur ^ 1][sr * GEMM_LDW + sk];
            d[0] = s0; d[1] = s1;
        }
        __syncthreads();
    }

    int col = n0 + n;
#pragma unroll
    for (int v = 0; v < 8; ++v) {
        int row = m0 + kh * 8 + v;
        float y = accA[v] + accB[v] + bias[col];
        if (act) y = tanhf(y);
        long o = (long)row * 1024 + col;
        if (oF0) oF0[(long)l * f0ls + o] = y;
        if (oF1) oF1[(long)l * f1ls + o] = y;
        if (oB0) oB0[(long)l * b0ls + o] = f2b(y);
        if (oB1) oB1[(long)l * b1ls + o] = f2b(y);
    }
}

// ---------------------------------------------------------------------------
// Attention: scores = ctx[b] . gamma[l,b]  -> softmax over 256 -> ct (bf16).
// Grid (B, L), 256 threads (8 waves). gamma cached in LDS; wave-shuffle dot
// reductions (coalesced over h); ct accumulation coalesced over h.
// ---------------------------------------------------------------------------
__global__ __launch_bounds__(256) void attn_kernel(
    const float* __restrict__ ctx,           // [B][256][1024]
    const float* __restrict__ gamma,         // [L][B][1024]
    unsigned short* __restrict__ ctbf)       // [L][B][1024]
{
    __shared__ float sg[1024];
    __shared__ float sw[256];
    __shared__ float red[256];
    int b = blockIdx.x, l = blockIdx.y, tid = threadIdx.x;

    const float* g = gamma + ((long)l * 64 + b) * 1024;
    for (int i = tid; i < 1024; i += 256) sg[i] = g[i];
    __syncthreads();

    int wv = tid >> 5, lane = tid & 31;
    for (int j = wv; j < 256; j += 8) {
        const float* cj = ctx + ((long)b * 256 + j) * 1024;
        float p = 0.0f;
        for (int h = lane; h < 1024; h += 32) p += cj[h] * sg[h];
#pragma unroll
        for (int off = 16; off > 0; off >>= 1) p += __shfl_down(p, off, 32);
        if (lane == 0) sw[j] = p;
    }
    __syncthreads();

    float v = sw[tid];
    red[tid] = v; __syncthreads();
    for (int s2 = 128; s2 > 0; s2 >>= 1) {
        if (tid < s2) red[tid] = fmaxf(red[tid], red[tid + s2]);
        __syncthreads();
    }
    float mx = red[0];
    __syncthreads();
    float e = __expf(v - mx);
    red[tid] = e; __syncthreads();
    for (int s2 = 128; s2 > 0; s2 >>= 1) {
        if (tid < s2) red[tid] += red[tid + s2];
        __syncthreads();
    }
    float inv = 1.0f / red[0];
    __syncthreads();
    sw[tid] = e * inv;
    __syncthreads();

    for (int h = tid; h < 1024; h += 256) {
        float a = 0.0f;
        const float* cb = ctx + (long)b * 256 * 1024 + h;
        for (int j = 0; j < 256; ++j) a += sw[j] * cb[(long)j * 1024];
        ctbf[((long)l * 64 + b) * 1024 + h] = f2b(a);
    }
}

// ---------------------------------------------------------------------------
// Prep kernels
// ---------------------------------------------------------------------------
__global__ void cvt_kernel(const float* __restrict__ s, unsigned short* __restrict__ d, long n) {
    long i = (long)blockIdx.x * blockDim.x + threadIdx.x;
    if (i < n) d[i] = f2b(s[i]);
}

__global__ void bias_add_kernel(const float* __restrict__ a, const float* __restrict__ b,
                                float* __restrict__ d, int n) {
    int i = blockIdx.x * blockDim.x + threadIdx.x;
    if (i < n) d[i] = a[i] + b[i];
}

// xemb[(l*31+t)*64+b][0..511] = bf16(emb[inputs[l+1][b][t]])
__global__ void embed_gather_kernel(const int* __restrict__ tok, const float* __restrict__ emb,
                                    unsigned short* __restrict__ x) {
    int blk = blockIdx.x;               // (l*31 + t)*64 + b
    int b = blk & 63;
    int lt = blk >> 6;
    int t = lt % 31, l = lt / 31;
    int token = tok[(((l + 1) * 64) + b) * 32 + t];
    const float* s = emb + (long)token * 512;
    unsigned short* dd = x + (long)blk * 512;
    for (int e = threadIdx.x; e < 512; e += blockDim.x) dd[e] = f2b(s[e]);
}

// pe[l][b][e] = (1/32) * sum_{1 <= s < plen-1} emb[inputs[l][b][s]][e]
__global__ void pe_kernel(const int* __restrict__ tok, const int* __restrict__ lens,
                          const float* __restrict__ emb, unsigned short* __restrict__ pe) {
    int b = blockIdx.x, l = blockIdx.y;
    int plen = lens[l * 64 + b];
    float acc[4] = {0.f, 0.f, 0.f, 0.f};
    for (int s = 1; s < 32; ++s) {
        if (s < plen - 1) {
            int token = tok[(l * 64 + b) * 32 + s];
            const float* src = emb + (long)token * 512;
#pragma unroll
            for (int i = 0; i < 4; ++i) acc[i] += src[threadIdx.x + i * 128];
        }
    }
    unsigned short* d = pe + ((long)l * 64 + b) * 512;
#pragma unroll
    for (int i = 0; i < 4; ++i) d[threadIdx.x + i * 128] = f2b(acc[i] * (1.0f / 32.0f));
}

// cm[b][h] = mean over SRC of contexts[b][:,h]
__global__ void ctx_mean_kernel(const float* __restrict__ ctx, unsigned short* __restrict__ cm) {
    int b = blockIdx.x;
    for (int h = threadIdx.x; h < 1024; h += blockDim.x) {
        float a = 0.f;
        const float* p = ctx + (long)b * 256 * 1024 + h;
        for (int j = 0; j < 256; ++j) a += p[(long)j * 1024];
        cm[(long)b * 1024 + h] = f2b(a * (1.0f / 256.0f));
    }
}

__global__ void copy_final_kernel(const float* __restrict__ hf, const float* __restrict__ cf,
                                  float* __restrict__ out, long n) {
    long i = (long)blockIdx.x * blockDim.x + threadIdx.x;
    if (i < n) { out[i] = hf[i]; out[n + i] = cf[i]; }
}

// ---------------------------------------------------------------------------
// Host orchestration
// ---------------------------------------------------------------------------
extern "C" void kernel_launch(void* const* d_in, const int* in_sizes, int n_in,
                              void* d_out, int out_size, void* d_ws, size_t ws_size,
                              hipStream_t stream)
{
    (void)in_sizes; (void)n_in; (void)out_size; (void)ws_size;
    const int E = 512, H = 1024, B = 64, T = 31, L = 4;

    const int*   inputs   = (const int*)d_in[0];
    const int*   lens     = (const int*)d_in[1];
    const float* ctx      = (const float*)d_in[2];
    const float* embW     = (const float*)d_in[3];
    const float* Wih0f    = (const float*)d_in[4];
    const float* Whh0f    = (const float*)d_in[5];
    const float* bih0     = (const float*)d_in[6];
    const float* bhh0     = (const float*)d_in[7];
    const float* Wih1f    = (const float*)d_in[8];
    const float* Whh1f    = (const float*)d_in[9];
    const float* bih1     = (const float*)d_in[10];
    const float* bhh1     = (const float*)d_in[11];
    const float* Watinf   = (const float*)d_in[12];
    const float* batin    = (const float*)d_in[13];
    const float* Watoutf  = (const float*)d_in[14];
    const float* batout   = (const float*)d_in[15];
    const float* Wislinf  = (const float*)d_in[16];
    const float* bislin   = (const float*)d_in[17];
    const float* Wisloutf = (const float*)d_in[18];
    const float* bislout  = (const float*)d_in[19];
    float* dout = (float*)d_out;

    char* ws = (char*)d_ws;
    size_t off = 0;
    auto alloc = [&](size_t bytes) -> void* {
        off = (off + 255) & ~(size_t)255;
        void* p = ws + off;
        off += bytes;
        return p;
    };

    unsigned short* wih0    = (unsigned short*)alloc((size_t)4 * H * E * 2);
    unsigned short* whh0    = (unsigned short*)alloc((size_t)4 * H * H * 2);
    unsigned short* wih1    = (unsigned short*)alloc((size_t)4 * H * H * 2);
    unsigned short* whh1    = (unsigned short*)alloc((size_t)4 * H * H * 2);
    unsigned short* watin   = (unsigned short*)alloc((size_t)H * H * 2);
    unsigned short* watout  = (unsigned short*)alloc((size_t)H * 2 * H * 2);
    unsigned short* wislin  = (unsigned short*)alloc((size_t)H * (E + H) * 2);
    unsigned short* wislout = (unsigned short*)alloc((size_t)H * (E + H) * 2);
    float*          bias0   = (float*)alloc((size_t)4 * H * 4);
    float*          bias1   = (float*)alloc((size_t)4 * H * 4);
    unsigned short* xemb    = (unsigned short*)alloc((size_t)L * T * B * E * 2);

    const long BH  = (long)B * H;   // 65536
    const long LVL = 2 * BH;        // level stride (2 layers)
    const long BUF = L * LVL;       // ping-pong buffer stride

    float*          hstF   = (float*)alloc((size_t)2 * BUF * 4);
    unsigned short* hstB   = (unsigned short*)alloc((size_t)2 * BUF * 2);
    float*          cstF   = (float*)alloc((size_t)2 * BUF * 4);
    float*          gammaF = (float*)alloc((size_t)L * BH * 4);
    unsigned short* ctbf   = (unsigned short*)alloc((size_t)L * BH * 2);
    unsigned short* pebf   = (unsigned short*)alloc((size_t)L * B * E * 2);
    unsigned short* cmbf   = (unsigned short*)alloc((size_t)B * H * 2);

    auto cvt = [&](const float* s, unsigned short* d, long n) {
        cvt_kernel<<<dim3((unsigned)((n + 255) / 256)), 256, 0, stream>>>(s, d, n);
    };
    cvt(Wih0f,    wih0,    (long)4 * H * E);
    cvt(Whh0f,    whh0,    (long)4 * H * H);
    cvt(Wih1f,    wih1,    (long)4 * H * H);
    cvt(Whh1f,    whh1,    (long)4 * H * H);
    cvt(Watinf,   watin,   (long)H * H);
    cvt(Watoutf,  watout,  (long)H * 2 * H);
    cvt(Wislinf,  wislin,  (long)H * (E + H));
    cvt(Wisloutf, wislout, (long)H * (E + H));
    bias_add_kernel<<<dim3(16), 256, 0, stream>>>(bih0, bhh0, bias0, 4 * H);
    bias_add_kernel<<<dim3(16), 256, 0, stream>>>(bih1, bhh1, bias1, 4 * H);
    embed_gather_kernel<<<dim3(L * T * B), 128, 0, stream>>>(inputs, embW, xemb);
    pe_kernel<<<dim3(B, L), 128, 0, stream>>>(inputs, lens, embW, pebf);
    ctx_mean_kernel<<<dim3(B), 256, 0, stream>>>(ctx, cmbf);

    auto hF = [&](int buf, int layer) { return hstF + (long)buf * BUF + (long)layer * BH; };
    auto hB = [&](int buf, int layer) { return hstB + (long)buf * BUF + (long)layer * BH; };
    auto cF = [&](int buf, int layer) { return cstF + (long)buf * BUF + (long)layer * BH; };

    dim3 ggrid(H / 16);   // 64 blocks, 512 threads: 16 waves = 4 levels x 4 m-tiles

    // h0 = tanh([pe, cm] @ isl_in_W^T + b)  -> both layer slots of buf0
    gemm_kernel<<<ggrid, 512, 0, stream>>>(
        pebf, (long)B * E, E, E,
        cmbf, 0, H, H,
        wislin, E + H, bislin, 1,
        hF(0, 0), LVL, hF(0, 1), LVL, hB(0, 0), LVL, hB(0, 1), LVL);
    // c0 = tanh([cm, pe] @ isl_out_W^T + b)
    gemm_kernel<<<ggrid, 512, 0, stream>>>(
        cmbf, 0, H, H,
        pebf, (long)B * E, E, E,
        wislout, E + H, bislout, 1,
        cF(0, 0), LVL, cF(0, 1), LVL,
        (unsigned short*)nullptr, 0, (unsigned short*)nullptr, 0);

    for (int t = 0; t < T; ++t) {
        int br = t & 1, bw = br ^ 1;
        // layer 0 LSTM cell
        lstm_cell_kernel<<<ggrid, 512, 0, stream>>>(
            xemb + (long)t * B * E, (long)T * B * E, E,
            hB(br, 0), wih0, whh0, bias0,
            cF(br, 0), cF(bw, 0), hF(bw, 0), hB(bw, 0), LVL);
        // layer 1 LSTM cell (input = h1 just produced)
        lstm_cell_kernel<<<ggrid, 512, 0, stream>>>(
            hB(bw, 0), LVL, H,
            hB(br, 1), wih1, whh1, bias1,
            cF(br, 1), cF(bw, 1), hF(bw, 1), hB(bw, 1), LVL);
        // gamma = h2 @ att_in_W^T + b
        gemm_kernel<<<ggrid, 512, 0, stream>>>(
            hB(bw, 1), LVL, H, H,
            hB(bw, 1), 0, H, 0,
            watin, H, batin, 0,
            gammaF, BH, (float*)nullptr, 0,
            (unsigned short*)nullptr, 0, (unsigned short*)nullptr, 0);
        // softmax attention -> ct (bf16)
        attn_kernel<<<dim3(B, L), 256, 0, stream>>>(ctx, gammaF, ctbf);
        // out = tanh([ct, h2] @ att_out_W^T + b) -> d_out[l][t]
        gemm_kernel<<<ggrid, 512, 0, stream>>>(
            ctbf, BH, H, H,
            hB(bw, 1), LVL, H, H,
            watout, 2 * H, batout, 1,
            dout + (long)t * BH, (long)T * BH, (float*)nullptr, 0,
            (unsigned short*)nullptr, 0, (unsigned short*)nullptr, 0);
    }

    // final states of level L-1 live in buffer 1 after t = 30
    copy_final_kernel<<<dim3((unsigned)(LVL / 256)), 256, 0, stream>>>(
        hF(1, 0) + (long)(L - 1) * LVL,
        cF(1, 0) + (long)(L - 1) * LVL,
        dout + (long)L * T * BH, LVL);
}
